// LSTMGenerator_25486335934677
// MI455X (gfx1250) — matrix operand order repaired and checked by reading, not compiled, not run on hardware
//
#include <hip/hip_runtime.h>
#include <hip/hip_bf16.h>
#include <cstddef>

// ---------------------------------------------------------------------------
// Sizes from the reference
// ---------------------------------------------------------------------------
#define BATCH 512
#define RROI  36
#define IMGD  2048
#define HID   768
#define V1    10001
#define VPAD  10016     // V1 rounded up to 16
#define NSTEP 12

typedef __attribute__((ext_vector_type(16))) __bf16 v16bf;
typedef __attribute__((ext_vector_type(8)))  __bf16 v8bf;
typedef __attribute__((ext_vector_type(8)))  float  v8f;

// d_in flat order: setup_inputs() insertion order, params flattened depth-first
enum {
  IN_IMG = 0, IN_CLS = 1, IN_VF = 2,
  IN_EMB = 3,
  IN_ATT_Q_W = 4,  IN_ATT_Q_B = 5,
  IN_ATT_V_W = 6,  IN_ATT_V_B = 7,
  IN_ATT_H_W = 8,  IN_ATT_H_B = 9,
  IN_ATT_W   = 10, IN_ATT_B   = 11,
  IN_Q_FC_W  = 12, IN_Q_FC_B  = 13,
  IN_V_FC_W  = 14, IN_V_FC_B  = 15,
  IN_LANG_FC_W = 16, IN_LANG_FC_B = 17,
  IN_GATE_W  = 18, IN_GATE_B  = 19,
  IN_A_UX_W = 20, IN_A_UX_B = 21, IN_A_UH_W = 22, IN_A_UH_B = 23,
  IN_A_RX_W = 24, IN_A_RX_B = 25, IN_A_RH_W = 26, IN_A_RH_B = 27,
  IN_A_MX_W = 28, IN_A_MX_B = 29, IN_A_MH_W = 30, IN_A_MH_B = 31,
  IN_L_UX_W = 32, IN_L_UX_B = 33, IN_L_UH_W = 34, IN_L_UH_B = 35,
  IN_L_RX_W = 36, IN_L_RX_B = 37, IN_L_RH_W = 38, IN_L_RH_B = 39,
  IN_L_MX_W = 40, IN_L_MX_B = 41, IN_L_MH_W = 42, IN_L_MH_B = 43,
  IN_SMAP = 44
};

// ---------------------------------------------------------------------------
// Weight pre-pack: fp32 row-major [K,N] -> bf16 WMMA B-fragment tiles.
// B fragment (32x16 K x N, wave32): lane = n_in_tile + 16*(k>=16 half),
// element e = k & 15. Tile index = n_tile * (Kd/32) + k_tile.
// Supports placing a source block at (k0,n0) inside a larger stacked matrix
// (fuses the GRU gate weights [ux;uh] x [u|r] into one GEMM operand).
// ---------------------------------------------------------------------------
__global__ __launch_bounds__(256)
void pack_w_kernel(const float* __restrict__ W, int K, int N,
                   __bf16* __restrict__ dst, int k0, int n0, int Kd, int Ndp)
{
  int i = blockIdx.x * 256 + threadIdx.x;
  if (i >= K * N) return;
  int k = i / N, n = i % N;
  int kg = k0 + k, ng = n0 + n;
  int ktD  = Kd >> 5;
  int kt   = kg >> 5, kin = kg & 31;
  int nt   = ng >> 4, nin = ng & 15;
  int lane = nin + ((kin & 16) ? 16 : 0);
  int e    = kin & 15;
  size_t idx = (((size_t)nt * ktD + kt) * 32 + lane) * 16 + e;
  dst[idx] = (__bf16)W[i];
  (void)Ndp;
}

__global__ __launch_bounds__(256)
void add_vec2_kernel(float* __restrict__ dst, const float* __restrict__ a,
                     const float* __restrict__ b, int n)
{
  int i = blockIdx.x * 256 + threadIdx.x;
  if (i < n) dst[i] = a[i] + b[i];
}

// Build a 16-element A fragment from a bf16 LDS row:
// lane<16  -> K 0..15 at +0
// lane>=16 -> K 16..31 at +16
static __device__ inline v16bf load_afrag(const __bf16* rowp, int lane)
{
  int o = (lane < 16) ? 0 : 16;
  return *(const v16bf*)(rowp + o);
}

// ---------------------------------------------------------------------------
// Generic WMMA GEMM: C[M,N] = epilogue(A[M,K](fp32) @ Wpacked(bf16) + bias)
// Block = 256 threads = 8 waves. Each wave: 32(M) x 16(N) tile. WG tile =
// 32 x 128. A is staged in LDS *already converted to bf16* (4 v_cvt_pk per
// thread per stage), so fragment build is pure ds_load_b128 + shuffle.
// K unrolled x2: one barrier pair covers 4 WMMAs. K must be a multiple of
// 64, M a multiple of 32 (true at every call site). The N-guard is
// wave-uniform so EXEC is all-ones at every WMMA (ISA requirement).
// act: 0=none 1=tanh 2=sigmoid. mul: optional rowwise Hadamard
// C[m,n] *= mul[(m/mulDiv)*N + n]  (q_att / q_enc fusions, N==768 there).
// ---------------------------------------------------------------------------
__global__ __launch_bounds__(256)
void gemm_wmma_kernel(const float* __restrict__ A, int lda,
                      const __bf16* __restrict__ Wp,
                      const float* __restrict__ bias,
                      float* __restrict__ C, int ldc,
                      int M, int K, int N, int Npad,
                      int act, const float* __restrict__ mul, int mulDiv)
{
  __shared__ __bf16 Asb[32][64];               // 4KB bf16 A stage (32M x 64K)
  const int lane  = threadIdx.x & 31;
  const int wave  = threadIdx.x >> 5;
  const int mBase = blockIdx.y * 32;
  const int nBase = blockIdx.x * 128 + wave * 16;
  const int kt    = K >> 5;                    // 32-deep B tiles
  const int kt2   = K >> 6;                    // 64-deep stages
  const bool nValid = (nBase < Npad);          // uniform per wave
  const size_t nTileBase = (size_t)(nValid ? (nBase >> 4) : 0) * kt;
  const int mm = lane & 15;

  const int tIdx  = threadIdx.x * 8;           // 8 floats per thread per stage
  const int ldRow = tIdx >> 6, ldCol = tIdx & 63;

  v8f acc0 = {}; v8f acc1 = {};

  for (int k2 = 0; k2 < kt2; ++k2) {
    // cooperative load 32x64 fp32, convert to bf16, store to LDS
    const float* src = A + (size_t)(mBase + ldRow) * lda + (size_t)k2 * 64 + ldCol;
    float4 f0 = *(const float4*)src;
    float4 f1 = *(const float4*)(src + 4);
    v8bf pk;
    pk[0] = (__bf16)f0.x; pk[1] = (__bf16)f0.y;
    pk[2] = (__bf16)f0.z; pk[3] = (__bf16)f0.w;
    pk[4] = (__bf16)f1.x; pk[5] = (__bf16)f1.y;
    pk[6] = (__bf16)f1.z; pk[7] = (__bf16)f1.w;
    *(v8bf*)&Asb[ldRow][ldCol] = pk;
    __syncthreads();

#pragma unroll
    for (int s = 0; s < 2; ++s) {
      v16bf a0 = load_afrag(&Asb[mm][s * 32], lane);
      v16bf a1 = load_afrag(&Asb[mm + 16][s * 32], lane);
      const __bf16* bp = Wp + ((nTileBase + (k2 * 2 + s)) * 32 + lane) * 16;
      v16bf bfrag = *(const v16bf*)bp;
      acc0 = __builtin_amdgcn_wmma_f32_16x16x32_bf16(
          false, a0, false, bfrag, (short)0, acc0, false, false);
      acc1 = __builtin_amdgcn_wmma_f32_16x16x32_bf16(
          false, a1, false, bfrag, (short)0, acc1, false, false);
    }
    __syncthreads();
  }

  if (!nValid) return;
  int nCol = nBase + mm;
  if (nCol >= N) return;                       // drop padded columns
  float bv  = bias ? bias[nCol] : 0.f;
  int mOff  = (lane < 16) ? 0 : 8;             // C layout: vgpr j -> M = mOff+j
#pragma unroll
  for (int j = 0; j < 8; ++j) {
    {
      int mRow = mBase + mOff + j;
      float x = acc0[j] + bv;
      if (act == 1) x = tanhf(x);
      else if (act == 2) x = 1.f / (1.f + __expf(-x));
      if (mul) x *= mul[(size_t)(mRow / mulDiv) * N + nCol];
      C[(size_t)mRow * ldc + nCol] = x;
    }
    {
      int mRow = mBase + 16 + mOff + j;
      float x = acc1[j] + bv;
      if (act == 1) x = tanhf(x);
      else if (act == 2) x = 1.f / (1.f + __expf(-x));
      if (mul) x *= mul[(size_t)(mRow / mulDiv) * N + nCol];
      C[(size_t)mRow * ldc + nCol] = x;
    }
  }
  (void)M;
}

// ---------------------------------------------------------------------------
// Elementwise / small kernels
// ---------------------------------------------------------------------------
__global__ __launch_bounds__(256)
void concat_rows_kernel(float* __restrict__ dst, const float* s0, const float* s1,
                        const float* s2, const float* s3, int nseg, int total)
{
  int i = blockIdx.x * 256 + threadIdx.x;
  if (i >= total) return;
  int colw = nseg * HID;
  int b = i / colw, col = i % colw;
  int seg = col / HID, c = col % HID;
  const float* s = (seg == 0) ? s0 : (seg == 1) ? s1 : (seg == 2) ? s2 : s3;
  dst[i] = s[(size_t)b * HID + c];
}

// cat[:, off:off+HID] = r * h   (r = zr[:, HID:2*HID])
__global__ __launch_bounds__(256)
void rh_overwrite_kernel(float* __restrict__ cat, int ldcat, int off,
                         const float* __restrict__ zr, const float* __restrict__ h)
{
  int i = blockIdx.x * 256 + threadIdx.x;
  if (i >= BATCH * HID) return;
  int b = i / HID, c = i % HID;
  cat[(size_t)b * ldcat + off + c] = zr[(size_t)b * (2 * HID) + HID + c] * h[i];
}

// hout = (1-z)*hin + z*m   (z = zr[:, 0:HID])
__global__ __launch_bounds__(256)
void gru_combine_kernel(float* __restrict__ hout, const float* __restrict__ zr,
                        const float* __restrict__ m, const float* __restrict__ hin)
{
  int i = blockIdx.x * 256 + threadIdx.x;
  if (i >= BATCH * HID) return;
  int b = i / HID;
  float z = zr[(size_t)b * (2 * HID) + (i % HID)];
  hout[i] = (1.f - z) * hin[i] + z * m[i];
}

__global__ __launch_bounds__(256)
void fuse_reduce_kernel(const float* __restrict__ fuse,
                        float* __restrict__ fsum, float* __restrict__ fmean)
{
  int i = blockIdx.x * 256 + threadIdx.x;
  if (i >= BATCH * HID) return;
  int b = i / HID, c = i % HID;
  float s = 0.f;
  for (int r = 0; r < RROI; ++r) s += fuse[((size_t)b * RROI + r) * HID + c];
  fsum[i] = s; fmean[i] = s * (1.f / RROI);
}

// One block per batch row: attention scores over R, softmax, weighted img sum.
__global__ __launch_bounds__(256)
void attention_kernel(const float* __restrict__ hatt, const float* __restrict__ fuse,
                      const float* __restrict__ attw, const float* __restrict__ attb,
                      const float* __restrict__ img, float* __restrict__ attx)
{
  int b = blockIdx.x;
  int lane = threadIdx.x & 31, wave = threadIdx.x >> 5;
  __shared__ float sc[RROI];
  const float* hrow = hatt + (size_t)b * HID;
  for (int r = wave; r < RROI; r += 8) {
    const float* frow = fuse + ((size_t)b * RROI + r) * HID;
    float p = 0.f;
    for (int c = lane; c < HID; c += 32) p += tanhf(hrow[c] + frow[c]) * attw[c];
    for (int o = 16; o > 0; o >>= 1) p += __shfl_down(p, o, 32);
    if (lane == 0) sc[r] = p + attb[0];
  }
  __syncthreads();
  float mx = -3.4e38f;
  for (int r = 0; r < RROI; ++r) mx = fmaxf(mx, sc[r]);
  float sum = 0.f;
  for (int r = 0; r < RROI; ++r) sum += __expf(sc[r] - mx);
  int t = threadIdx.x;
  float e = (t < RROI) ? __expf(sc[t] - mx) / sum : 0.f;
  __syncthreads();
  if (t < RROI) sc[t] = e;
  __syncthreads();
  const float* irow = img + (size_t)b * RROI * IMGD;
  for (int c = t; c < IMGD; c += 256) {
    float a = 0.f;
    for (int r = 0; r < RROI; ++r) a += sc[r] * irow[(size_t)r * IMGD + c];
    attx[(size_t)b * IMGD + c] = a;
  }
}

// One block per batch row: sim = softmax_r(h2 . visual_feat), gate = sigmoid(h2 . gate_w + b)
__global__ __launch_bounds__(256)
void simgate_kernel(const float* __restrict__ h2, const float* __restrict__ vf,
                    const float* __restrict__ gw, const float* __restrict__ gb,
                    float* __restrict__ sim, float* __restrict__ gatev)
{
  int b = blockIdx.x;
  int lane = threadIdx.x & 31, wave = threadIdx.x >> 5;
  __shared__ float sc[RROI + 1];
  const float* hrow = h2 + (size_t)b * HID;
  for (int r = wave; r <= RROI; r += 8) {
    const float* w = (r < RROI) ? (vf + ((size_t)b * RROI + r) * HID) : gw;
    float p = 0.f;
    for (int c = lane; c < HID; c += 32) p += hrow[c] * w[c];
    for (int o = 16; o > 0; o >>= 1) p += __shfl_down(p, o, 32);
    if (lane == 0) sc[r] = p;
  }
  __syncthreads();
  float mx = -3.4e38f;
  for (int r = 0; r < RROI; ++r) mx = fmaxf(mx, sc[r]);
  float sum = 0.f;
  for (int r = 0; r < RROI; ++r) sum += __expf(sc[r] - mx);
  int t = threadIdx.x;
  if (t < RROI) sim[(size_t)b * RROI + t] = __expf(sc[t] - mx) / sum;
  if (t == 0)   gatev[b] = 1.f / (1.f + __expf(-(sc[RROI] + gb[0])));
}

// One block per batch row: softmax over logits, gate-mix with sim @ SM^T,
// write output, argmax, gather next-word embedding into prev.
__global__ __launch_bounds__(256)
void combine_kernel(const float* __restrict__ logits, int ldl,
                    const float* __restrict__ sim, const float* __restrict__ gatev,
                    const float* __restrict__ SM, const float* __restrict__ embed,
                    float* __restrict__ out_sent, float* __restrict__ out_gate,
                    float* __restrict__ prev, int step)
{
  int b = blockIdx.x, t = threadIdx.x;
  __shared__ float red[256];
  __shared__ int   redi[256];
  __shared__ float s_mx, s_sum;
  __shared__ int   s_idx;
  const float* lg = logits + (size_t)b * ldl;

  float mx = -3.4e38f;
  for (int v = t; v < V1; v += 256) mx = fmaxf(mx, lg[v]);
  red[t] = mx; __syncthreads();
  for (int o = 128; o > 0; o >>= 1) { if (t < o) red[t] = fmaxf(red[t], red[t + o]); __syncthreads(); }
  if (t == 0) s_mx = red[0]; __syncthreads();
  mx = s_mx;

  float sm = 0.f;
  for (int v = t; v < V1; v += 256) sm += __expf(lg[v] - mx);
  red[t] = sm; __syncthreads();
  for (int o = 128; o > 0; o >>= 1) { if (t < o) red[t] += red[t + o]; __syncthreads(); }
  if (t == 0) s_sum = red[0]; __syncthreads();
  float inv = 1.f / s_sum;

  float g = gatev[b];
  float simr[RROI];
#pragma unroll
  for (int r = 0; r < RROI; ++r) simr[r] = sim[(size_t)b * RROI + r];

  float bestv = -3.4e38f; int besti = 0;
  float* outp = out_sent + ((size_t)b * NSTEP + step) * V1;
  for (int v = t; v < V1; v += 256) {
    float pred = __expf(lg[v] - mx) * inv;
    const float* smrow = SM + (size_t)v * RROI;
    float sp = 0.f;
#pragma unroll
    for (int r = 0; r < RROI; ++r) sp += simr[r] * smrow[r];
    float val = g * pred + (1.f - g) * sp;
    outp[v] = val;
    if (val > bestv) { bestv = val; besti = v; }
  }
  red[t] = bestv; redi[t] = besti; __syncthreads();
  for (int o = 128; o > 0; o >>= 1) {
    if (t < o) {
      if (red[t + o] > red[t] || (red[t + o] == red[t] && redi[t + o] < redi[t])) {
        red[t] = red[t + o]; redi[t] = redi[t + o];
      }
    }
    __syncthreads();
  }
  if (t == 0) { s_idx = redi[0]; out_gate[(size_t)b * NSTEP + step] = g; }
  __syncthreads();
  int idx = s_idx;
  for (int c = t; c < HID; c += 256)
    prev[(size_t)b * HID + c] = embed[(size_t)idx * HID + c];
}

// ---------------------------------------------------------------------------
// Host orchestration
// ---------------------------------------------------------------------------
extern "C" void kernel_launch(void* const* d_in, const int* in_sizes, int n_in,
                              void* d_out, int out_size, void* d_ws, size_t ws_size,
                              hipStream_t stream)
{
  (void)in_sizes; (void)n_in; (void)out_size; (void)ws_size;
  const float* img  = (const float*)d_in[IN_IMG];
  const float* cls  = (const float*)d_in[IN_CLS];
  const float* vf   = (const float*)d_in[IN_VF];
  const float* emb  = (const float*)d_in[IN_EMB];
  const float* SM   = (const float*)d_in[IN_SMAP];

  // workspace bump allocator (256B aligned)
  size_t off = 0;
  auto alloc = [&](size_t bytes) -> void* {
    void* p = (char*)d_ws + off;
    off += (bytes + 255) & ~(size_t)255;
    return p;
  };
  auto fbuf = [&](size_t n) -> float* { return (float*)alloc(n * sizeof(float)); };
  auto wbuf = [&](size_t k, size_t n) -> __bf16* { return (__bf16*)alloc(k * n * sizeof(__bf16)); };

  __bf16* pw_att_q   = wbuf(HID, HID);
  __bf16* pw_q_fc    = wbuf(HID, HID);
  __bf16* pw_att_h   = wbuf(HID, HID);
  __bf16* pw_att_v   = wbuf(IMGD, HID);
  __bf16* pw_v_fc    = wbuf(IMGD, HID);
  __bf16* pw_zr_att  = wbuf(3072, 1536);
  __bf16* pw_m_att   = wbuf(3072, HID);
  __bf16* pw_zr_lang = wbuf(2304, 1536);
  __bf16* pw_m_lang  = wbuf(2304, HID);
  __bf16* pw_lang    = wbuf(HID, VPAD);

  float* b_zr_att  = fbuf(1536);
  float* b_m_att   = fbuf(HID);
  float* b_zr_lang = fbuf(1536);
  float* b_m_lang  = fbuf(HID);

  float* q_att     = fbuf((size_t)BATCH * HID);
  float* q_enc     = fbuf((size_t)BATCH * HID);
  float* fuse      = fbuf((size_t)BATCH * RROI * HID);
  float* fsum      = fbuf((size_t)BATCH * HID);
  float* fmean     = fbuf((size_t)BATCH * HID);
  float* cat1      = fbuf((size_t)BATCH * 3072);
  float* cat2      = fbuf((size_t)BATCH * 2304);
  float* zr1       = fbuf((size_t)BATCH * 1536);
  float* zr2       = fbuf((size_t)BATCH * 1536);
  float* m1        = fbuf((size_t)BATCH * HID);
  float* m2        = fbuf((size_t)BATCH * HID);
  float* h1        = fbuf((size_t)BATCH * HID);
  float* h2        = fbuf((size_t)BATCH * HID);
  float* hatt      = fbuf((size_t)BATCH * HID);
  float* fenc      = fbuf((size_t)BATCH * HID);
  float* prev      = fbuf((size_t)BATCH * HID);
  float* attx      = fbuf((size_t)BATCH * IMGD);
  float* logits    = fbuf((size_t)BATCH * VPAD);
  float* sim       = fbuf((size_t)BATCH * RROI);
  float* gatev     = fbuf(BATCH);

  float* out_sent = (float*)d_out;
  float* out_gate = out_sent + (size_t)BATCH * NSTEP * V1;

  auto pk = [&](int in_idx, int K, int N, __bf16* dst, int k0, int n0, int Kd, int Ndp) {
    int tot = K * N;
    pack_w_kernel<<<(tot + 255) / 256, 256, 0, stream>>>(
        (const float*)d_in[in_idx], K, N, dst, k0, n0, Kd, Ndp);
  };
  auto gemm = [&](const float* A, int lda, const __bf16* Wp, const float* bias,
                  float* C, int ldc, int M, int K, int N, int Npad,
                  int act, const float* mul, int mulDiv) {
    dim3 g((Npad + 127) / 128, M / 32);
    gemm_wmma_kernel<<<g, 256, 0, stream>>>(A, lda, Wp, bias, C, ldc,
                                            M, K, N, Npad, act, mul, mulDiv);
  };
  auto addv = [&](float* dst, int ia, int ib, int n) {
    add_vec2_kernel<<<(n + 255) / 256, 256, 0, stream>>>(
        dst, (const float*)d_in[ia], (const float*)d_in[ib], n);
  };

  // ---- pack weights (fp32 -> bf16 WMMA B-fragments) ----
  pk(IN_ATT_Q_W, HID, HID, pw_att_q, 0, 0, HID, HID);
  pk(IN_Q_FC_W,  HID, HID, pw_q_fc,  0, 0, HID, HID);
  pk(IN_ATT_H_W, HID, HID, pw_att_h, 0, 0, HID, HID);
  pk(IN_ATT_V_W, IMGD, HID, pw_att_v, 0, 0, IMGD, HID);
  pk(IN_V_FC_W,  IMGD, HID, pw_v_fc,  0, 0, IMGD, HID);
  // att_rnn: [ [ux|rx] ; [uh|rh] ]  (K=3072, N=1536),  m: [mx;mh] (K=3072,N=768)
  pk(IN_A_UX_W, 2304, HID, pw_zr_att, 0,    0,   3072, 1536);
  pk(IN_A_RX_W, 2304, HID, pw_zr_att, 0,    HID, 3072, 1536);
  pk(IN_A_UH_W, HID,  HID, pw_zr_att, 2304, 0,   3072, 1536);
  pk(IN_A_RH_W, HID,  HID, pw_zr_att, 2304, HID, 3072, 1536);
  pk(IN_A_MX_W, 2304, HID, pw_m_att,  0,    0,   3072, HID);
  pk(IN_A_MH_W, HID,  HID, pw_m_att,  2304, 0,   3072, HID);
  // lang_rnn: K=2304
  pk(IN_L_UX_W, 1536, HID, pw_zr_lang, 0,    0,   2304, 1536);
  pk(IN_L_RX_W, 1536, HID, pw_zr_lang, 0,    HID, 2304, 1536);
  pk(IN_L_UH_W, HID,  HID, pw_zr_lang, 1536, 0,   2304, 1536);
  pk(IN_L_RH_W, HID,  HID, pw_zr_lang, 1536, HID, 2304, 1536);
  pk(IN_L_MX_W, 1536, HID, pw_m_lang,  0,    0,   2304, HID);
  pk(IN_L_MH_W, HID,  HID, pw_m_lang,  1536, 0,   2304, HID);
  pk(IN_LANG_FC_W, HID, V1, pw_lang, 0, 0, HID, VPAD);

  // ---- combined GRU biases (x-bias + h-bias) ----
  addv(b_zr_att,        IN_A_UX_B, IN_A_UH_B, HID);
  addv(b_zr_att + HID,  IN_A_RX_B, IN_A_RH_B, HID);
  addv(b_m_att,         IN_A_MX_B, IN_A_MH_B, HID);
  addv(b_zr_lang,       IN_L_UX_B, IN_L_UH_B, HID);
  addv(b_zr_lang + HID, IN_L_RX_B, IN_L_RH_B, HID);
  addv(b_m_lang,        IN_L_MX_B, IN_L_MH_B, HID);

  // ---- precompute ----
  gemm(cls, HID, pw_att_q, (const float*)d_in[IN_ATT_Q_B], q_att, HID,
       BATCH, HID, HID, HID, /*tanh*/1, nullptr, 1);
  gemm(cls, HID, pw_q_fc, (const float*)d_in[IN_Q_FC_B], q_enc, HID,
       BATCH, HID, HID, HID, /*tanh*/1, nullptr, 1);
  // fuse_feat = tanh(img @ att_v_w + b) * q_att[b]  (rows = b*R+r, mulDiv=R)
  gemm(img, IMGD, pw_att_v, (const float*)d_in[IN_ATT_V_B], fuse, HID,
       BATCH * RROI, IMGD, HID, HID, /*tanh*/1, q_att, RROI);
  {
    int tot = BATCH * HID;
    fuse_reduce_kernel<<<(tot + 255) / 256, 256, 0, stream>>>(fuse, fsum, fmean);
  }
  hipMemsetAsync(h2,   0, (size_t)BATCH * HID * sizeof(float), stream);
  hipMemsetAsync(prev, 0, (size_t)BATCH * HID * sizeof(float), stream);

  // ---- 12 decode steps ----
  for (int step = 0; step < NSTEP; ++step) {
    const float* fu = (step == 0) ? fmean : fsum;
    // cat1 = [fuse, h2, prev, h2]
    {
      int tot = BATCH * 4 * HID;
      concat_rows_kernel<<<(tot + 255) / 256, 256, 0, stream>>>(
          cat1, fu, h2, prev, h2, 4, tot);
    }
    gemm(cat1, 3072, pw_zr_att, b_zr_att, zr1, 1536,
         BATCH, 3072, 1536, 1536, /*sigmoid*/2, nullptr, 1);
    {
      int tot = BATCH * HID;
      rh_overwrite_kernel<<<(tot + 255) / 256, 256, 0, stream>>>(cat1, 3072, 2304, zr1, h2);
    }
    gemm(cat1, 3072, pw_m_att, b_m_att, m1, HID,
         BATCH, 3072, HID, HID, /*tanh*/1, nullptr, 1);
    {
      int tot = BATCH * HID;
      gru_combine_kernel<<<(tot + 255) / 256, 256, 0, stream>>>(h1, zr1, m1, h2);
    }
    gemm(h1, HID, pw_att_h, (const float*)d_in[IN_ATT_H_B], hatt, HID,
         BATCH, HID, HID, HID, /*none*/0, nullptr, 1);
    attention_kernel<<<BATCH, 256, 0, stream>>>(
        hatt, fuse, (const float*)d_in[IN_ATT_W], (const float*)d_in[IN_ATT_B], img, attx);
    // fuse_enc = tanh(att_x @ v_fc + b) * q_enc
    gemm(attx, IMGD, pw_v_fc, (const float*)d_in[IN_V_FC_B], fenc, HID,
         BATCH, IMGD, HID, HID, /*tanh*/1, q_enc, 1);
    // cat2 = [fuse_enc, h1, h2]
    {
      int tot = BATCH * 3 * HID;
      concat_rows_kernel<<<(tot + 255) / 256, 256, 0, stream>>>(
          cat2, fenc, h1, h2, h2, 3, tot);
    }
    gemm(cat2, 2304, pw_zr_lang, b_zr_lang, zr2, 1536,
         BATCH, 2304, 1536, 1536, /*sigmoid*/2, nullptr, 1);
    {
      int tot = BATCH * HID;
      rh_overwrite_kernel<<<(tot + 255) / 256, 256, 0, stream>>>(cat2, 2304, 1536, zr2, h2);
    }
    gemm(cat2, 2304, pw_m_lang, b_m_lang, m2, HID,
         BATCH, 2304, HID, HID, /*tanh*/1, nullptr, 1);
    {
      int tot = BATCH * HID;
      gru_combine_kernel<<<(tot + 255) / 256, 256, 0, stream>>>(h2, zr2, m2, h2);
    }
    gemm(h2, HID, pw_lang, (const float*)d_in[IN_LANG_FC_B], logits, VPAD,
         BATCH, HID, V1, VPAD, /*none*/0, nullptr, 1);
    simgate_kernel<<<BATCH, 256, 0, stream>>>(
        h2, vf, (const float*)d_in[IN_GATE_W], (const float*)d_in[IN_GATE_B], sim, gatev);
    combine_kernel<<<BATCH, 256, 0, stream>>>(
        logits, VPAD, sim, gatev, SM, emb, out_sent, out_gate, prev, step);
  }
}
